// ResidualVectorQuantizer_84318797955168
// MI455X (gfx1250) — matrix operand order repaired and checked by reading, not compiled, not run on hardware
//
#include <hip/hip_runtime.h>
#include <hip/hip_bf16.h>

// ---------------- problem constants ----------------
#define DEPTH   8
#define KCB     1024
#define DIM     256
#define NTOK    131072

#define MT      128     // tokens per workgroup tile
#define RS      260     // LDS residual row stride in floats (pad: kills bank conflicts, keeps 16B align)

typedef __attribute__((ext_vector_type(16))) __bf16 v16bf;
typedef __attribute__((ext_vector_type(8)))  float  v8f;

// ---------------------------------------------------------------------------
// Pre-pass: convert codebooks f32 [DEPTH][KCB][DIM] into
//   (a) bf16 blob pre-swizzled into WMMA B-fragment layout:
//       blob[((level*64 + kt)*8 + chunk)*512 + lane*16 + i]
//       where lane<16 holds codeword n=kt*16+lane, elems i -> e = 32*chunk + i
//             lane>=16 holds same n, elems i -> e = 32*chunk + 16 + i
//   (b) bias[level][n] = -0.5 * ||c_n||^2   (argmax of r.c + bias == argmin dist)
// One block of 256 threads per (level, codeword) row.
// ---------------------------------------------------------------------------
__global__ void __launch_bounds__(256)
rvq_pack_kernel(const float* __restrict__ cb,
                unsigned short* __restrict__ blob,
                float* __restrict__ bias) {
    const int row   = blockIdx.x;        // 0 .. DEPTH*KCB-1
    const int level = row >> 10;
    const int n     = row & (KCB - 1);
    const int e     = threadIdx.x;       // 0..255

    const float v = cb[(size_t)row * DIM + e];

    // block reduction of v*v for the bias
    float s = v * v;
    #pragma unroll
    for (int m = 16; m >= 1; m >>= 1) s += __shfl_xor(s, m, 32);
    __shared__ float partial[8];
    if ((threadIdx.x & 31) == 0) partial[threadIdx.x >> 5] = s;
    __syncthreads();
    if (threadIdx.x == 0) {
        float t = 0.f;
        #pragma unroll
        for (int i = 0; i < 8; ++i) t += partial[i];
        bias[row] = -0.5f * t;
    }

    // scatter this element into the B-fragment swizzled blob (bf16)
    const int kt   = n >> 4;
    const int lc   = n & 15;
    const int c    = e >> 5;
    const int eo   = e & 31;
    const int lane = lc + ((eo >= 16) ? 16 : 0);
    const int i    = eo & 15;
    const size_t dst = ((((size_t)(level * 64 + kt)) * 8 + c) * 32 + lane) * 16 + i;
    blob[dst] = __builtin_bit_cast(unsigned short, (__bf16)v);
}

// ---------------------------------------------------------------------------
// Main residual-VQ kernel. 1024 blocks x 256 threads (8 waves of 32).
// Wave w owns token rows [16w, 16w+16) of the 128-token tile.
// Residual kept in f32 in LDS; per level:
//   1) each wave builds bf16 A fragments (16x32 per chunk, 8 chunks = D=256)
//   2) loop 64 k-tiles (2 at a time): 8 WMMA f32_16x16x32_bf16 per tile
//   3) per-token argmax (+bias) with xor-shuffle reduction over 16 lanes
//   4) subtract chosen codeword (f32, exact) from residual in LDS
// Final: out = latent - residual  (== quantized sum == straight-through fwd)
//
// __launch_bounds__(256, 1): LDS (130 KB) already caps us at 2 blocks/WGP,
// so let the compiler take the whole VGPR file — kills the A-fragment
// scratch spills observed in the round-1 inner loop.
// ---------------------------------------------------------------------------
__global__ void __launch_bounds__(256, 1)
rvq_main_kernel(const float* __restrict__ latent,
                const float* __restrict__ cb,
                const unsigned short* __restrict__ blob,
                const float* __restrict__ bias,
                float* __restrict__ outST,
                int* __restrict__ outCodes) {
    extern __shared__ float smem[];
    float* res  = smem;                       // [MT][RS] f32 residual
    int*   idxS = (int*)(smem + MT * RS);     // [MT] chosen code per token

    const int tid  = threadIdx.x;
    const int lane = tid & 31;
    const int wave = tid >> 5;
    const size_t tok0 = (size_t)blockIdx.x * MT;

    // ---- init: residual = latent (coalesced float4) ----
    {
        const float4* lat4 = (const float4*)(latent + tok0 * DIM);
        #pragma unroll
        for (int it = 0; it < (MT * DIM / 4) / 256; ++it) {
            int idx = it * 256 + tid;
            int t   = idx >> 6;        // 64 float4 per row
            int e4  = idx & 63;
            float4 v = lat4[idx];
            *(float4*)(res + t * RS + e4 * 4) = v;
        }
    }
    __syncthreads();

    const int mrow    = wave * 16 + (lane & 15);  // token row this lane represents
    const int halfsel = (lane >= 16) ? 16 : 0;    // A-fragment K-half select
    const int ncol    = lane & 15;                // codeword column within a k-tile

    for (int level = 0; level < DEPTH; ++level) {
        // ---- (1) build A fragments: residual row -> bf16, contiguous k = 16h + s layout (matches blob) ----
        v16bf A[8];
        #pragma unroll
        for (int c = 0; c < 8; ++c) {
            const float* rp = res + mrow * RS + c * 32 + halfsel;
            #pragma unroll
            for (int p = 0; p < 4; ++p) {
                float2 lo = *(const float2*)(rp + 2 * p);       // K = halfsel + 2p,2p+1
                float2 hi = *(const float2*)(rp + 8 + 2 * p);   // K = halfsel + 8 + 2p,...
                A[c][2 * p]     = (__bf16)lo.x;
                A[c][2 * p + 1] = (__bf16)lo.y;
                A[c][8 + 2 * p]     = (__bf16)hi.x;
                A[c][8 + 2 * p + 1] = (__bf16)hi.y;
            }
        }

        // ---- (2,3) score all 1024 codewords, track running argmax ----
        float bestS[8];
        int   bestI[8];
        #pragma unroll
        for (int j = 0; j < 8; ++j) { bestS[j] = -3.4e38f; bestI[j] = 0; }

        const unsigned short* Bbase = blob + (size_t)level * 64 * 8 * 32 * 16;
        const float*          biasL = bias + (size_t)level * KCB;

        for (int kt = 0; kt < 64; kt += 2) {
            v8f acc0 = {};
            v8f acc1 = {};
            const unsigned short* b0 = Bbase + ((size_t)kt * 8) * 512 + lane * 16;
            const unsigned short* b1 = b0 + 8 * 512;
            #pragma unroll
            for (int c = 0; c < 8; ++c) {
                v16bf B0 = *(const v16bf*)(b0 + c * 512);
                v16bf B1 = *(const v16bf*)(b1 + c * 512);
                acc0 = __builtin_amdgcn_wmma_f32_16x16x32_bf16(
                           false, A[c], false, B0, (short)0, acc0, false, false);
                acc1 = __builtin_amdgcn_wmma_f32_16x16x32_bf16(
                           false, A[c], false, B1, (short)0, acc1, false, false);
            }
            if (kt + 2 < 64) __builtin_prefetch(b0 + 2 * 8 * 512, 0, 0);

            const float bias0 = biasL[kt * 16 + ncol];
            const float bias1 = biasL[(kt + 1) * 16 + ncol];
            const int n0 = kt * 16 + ncol;
            const int n1 = n0 + 16;
            #pragma unroll
            for (int j = 0; j < 8; ++j) {
                float s0 = acc0[j] + bias0;
                float s1 = acc1[j] + bias1;
                if (s0 > bestS[j]) { bestS[j] = s0; bestI[j] = n0; }
                if (s1 > bestS[j]) { bestS[j] = s1; bestI[j] = n1; }
            }
        }

        // xor-shuffle argmax across the 16 lanes of each half-wave
        #pragma unroll
        for (int j = 0; j < 8; ++j) {
            float s  = bestS[j];
            int   bi = bestI[j];
            #pragma unroll
            for (int m = 8; m >= 1; m >>= 1) {
                float so = __shfl_xor(s, m, 32);
                int   io = __shfl_xor(bi, m, 32);
                if (so > s || (so == s && io < bi)) { s = so; bi = io; }
            }
            bestS[j] = s; bestI[j] = bi;
        }

        // lanes 0 and 16 publish winners for tokens wave*16+{0..7} / +{8..15}
        if (ncol == 0) {
            const int tloc = wave * 16 + ((lane >= 16) ? 8 : 0);
            #pragma unroll
            for (int j = 0; j < 8; ++j) {
                idxS[tloc + j] = bestI[j];
                outCodes[(size_t)level * NTOK + tok0 + tloc + j] = bestI[j];
            }
        }
        __syncthreads();

        // ---- (4) residual -= chosen codeword (exact f32 gather) ----
        {
            const int t    = tid & 127;   // token within tile
            const int h    = tid >> 7;    // embed half (0/1)
            const int code = idxS[t];
            const float4* crow = (const float4*)(cb + ((size_t)level * KCB + code) * DIM) + h * 32;
            float* rrow = res + t * RS + h * 128;
            #pragma unroll 4
            for (int q = 0; q < 32; ++q) {
                float4 cv = crow[q];
                float4 rv = *(float4*)(rrow + 4 * q);
                rv.x -= cv.x; rv.y -= cv.y; rv.z -= cv.z; rv.w -= cv.w;
                *(float4*)(rrow + 4 * q) = rv;
            }
        }
        __syncthreads();
    }

    // ---- out = latent - residual_final (== quantized sum) ----
    {
        const float4* lat4 = (const float4*)(latent + tok0 * DIM);
        float4*       out4 = (float4*)(outST + tok0 * DIM);
        #pragma unroll
        for (int it = 0; it < (MT * DIM / 4) / 256; ++it) {
            int idx = it * 256 + tid;
            int t   = idx >> 6;
            int e4  = idx & 63;
            float4 lv = lat4[idx];
            float4 rv = *(const float4*)(res + t * RS + e4 * 4);
            float4 ov;
            ov.x = lv.x - rv.x; ov.y = lv.y - rv.y;
            ov.z = lv.z - rv.z; ov.w = lv.w - rv.w;
            out4[idx] = ov;
        }
    }
}

// ---------------------------------------------------------------------------
extern "C" void kernel_launch(void* const* d_in, const int* in_sizes, int n_in,
                              void* d_out, int out_size, void* d_ws, size_t ws_size,
                              hipStream_t stream) {
    const float* latent = (const float*)d_in[0];   // [NTOK, DIM] f32
    const float* cb     = (const float*)d_in[1];   // [DEPTH, KCB, DIM] f32

    // workspace layout: bf16 B-blob (4 MB) then bias (32 KB)
    unsigned short* blob = (unsigned short*)d_ws;
    float*          bias = (float*)((char*)d_ws + (size_t)DEPTH * KCB * DIM * sizeof(unsigned short));

    float* outST    = (float*)d_out;                       // [NTOK, DIM] f32
    int*   outCodes = (int*)d_out + (size_t)NTOK * DIM;    // [DEPTH, NTOK] i32 bits

    rvq_pack_kernel<<<DEPTH * KCB, 256, 0, stream>>>(cb, blob, bias);

    const size_t shmem = (size_t)MT * RS * sizeof(float) + MT * sizeof(int);
    rvq_main_kernel<<<NTOK / MT, 256, shmem, stream>>>(latent, cb, blob, bias, outST, outCodes);
}
